// ATMAttention_36490042147463
// MI455X (gfx1250) — matrix operand order repaired and checked by reading, not compiled, not run on hardware
//
#include <hip/hip_runtime.h>
#include <hip/hip_bf16.h>

typedef __bf16 bf16;
typedef __attribute__((ext_vector_type(16))) __bf16 v16bf;
typedef __attribute__((ext_vector_type(8)))  float  v8f;
typedef unsigned int __attribute__((ext_vector_type(4))) su4;
typedef unsigned int __attribute__((ext_vector_type(8))) su8;

#define B_    8
#define NQ_   2048
#define NKV_  2048
#define C_    256
#define NH_   8
#define HD_   32
#define M_    1024      // (Hm/SR)*(Wm/SR) = 32*32
#define HW_   4096      // Hm*Wm
#define SCALE_f 0.17677669529663687f
#define INV_CNT (1.0f/1.000001f)
#define LN_EPS_ 1e-5f

__device__ __forceinline__ int lane_id() { return (int)(threadIdx.x & 31); }

union FragU { v16bf v; uint4 q[2]; };

// A fragment 16x32 bf16, row-major source. ISA layout: lanes 0-15 hold row=lane,
// K = [0..7] in elems 0..7 and [16..23] in elems 8..15; lanes 16-31: K=[8..15],[24..31].
__device__ __forceinline__ v16bf load_a16x32(const bf16* base, int ld) {
  int l = lane_id();
  const bf16* p = base + (l & 15) * ld + ((l & 16) ? 8 : 0);
  FragU r;
  r.q[0] = *(const uint4*)(p);
  r.q[1] = *(const uint4*)(p + 16);
  return r.v;
}

// B fragment 32x16 bf16 from Bt[N][K] row-major (i.e. B[k][n] = Bt[n][k]).
// Lane n in 0..15 holds column n, K=[0..7],[16..23]; lanes 16..31 hold K=[8..15],[24..31].
__device__ __forceinline__ v16bf load_bT16(const bf16* baseT, int ld) {
  int l = lane_id();
  const bf16* p = baseT + (l & 15) * ld + ((l & 16) ? 8 : 0);
  FragU r;
  r.q[0] = *(const uint4*)(p);
  r.q[1] = *(const uint4*)(p + 16);
  return r.v;
}

__device__ __forceinline__ v8f wmma_bf16(v16bf a, v16bf b, v8f c) {
  return __builtin_amdgcn_wmma_f32_16x16x32_bf16(false, a, false, b, (short)0, c, false, false);
}

// ---------------- conversion / packing ----------------
__global__ void k_cvt_bf16(const float* __restrict__ s, bf16* __restrict__ d, int n) {
  for (int i = blockIdx.x * blockDim.x + threadIdx.x; i < n; i += gridDim.x * blockDim.x)
    d[i] = (bf16)s[i];
}

// sr_w [co][ci][kh][kw] -> wt[co][(kh*2+kw)*256 + ci]  (bf16)
__global__ void k_pack_convw(const float* __restrict__ srw, bf16* __restrict__ wt) {
  int i = blockIdx.x * blockDim.x + threadIdx.x;
  if (i >= 256 * 1024) return;
  int co = i >> 10, k = i & 1023, pix = k >> 8, ci = k & 255;
  int kh = pix >> 1, kw = pix & 1;
  wt[i] = (bf16)srw[((co * 256 + ci) * 2 + kh) * 2 + kw];
}

// token2map gather fused with im2col: imcol[b][o][pix*256+ci] = kv_x[b][idx][ci]*INV
__global__ void k_im2col(const float* __restrict__ kvx, const int* __restrict__ idx,
                         bf16* __restrict__ imcol) {
  int t = blockIdx.x * blockDim.x + threadIdx.x;   // B*HW*64 (4 channels per thread)
  if (t >= B_ * HW_ * 64) return;
  int g = t & 63;
  int p = (t >> 6) & (HW_ - 1);
  int b = t >> 18;
  int h = p >> 6, w = p & 63;
  int o   = (h >> 1) * 32 + (w >> 1);
  int pix = ((h & 1) << 1) | (w & 1);
  long src = ((long)b * NKV_ + idx[b * HW_ + p]) * C_ + g * 4;
  float4 v = *(const float4*)(kvx + src);
  bf16* d = imcol + ((long)(b * M_ + o) * 1024 + pix * 256 + g * 4);
  d[0] = (bf16)(v.x * INV_CNT); d[1] = (bf16)(v.y * INV_CNT);
  d[2] = (bf16)(v.z * INV_CNT); d[3] = (bf16)(v.w * INV_CNT);
}

// conf[b][o] = mean of 4 gathered token scores
__global__ void k_conf(const float* __restrict__ score, const int* __restrict__ idx,
                       float* __restrict__ conf) {
  int t = blockIdx.x * blockDim.x + threadIdx.x;
  if (t >= B_ * M_) return;
  int b = t >> 10, o = t & 1023;
  int oh = o >> 5, ow = o & 31;
  float s = 0.f;
  #pragma unroll
  for (int kh = 0; kh < 2; ++kh)
    #pragma unroll
    for (int kw = 0; kw < 2; ++kw) {
      int p = (2 * oh + kh) * 64 + 2 * ow + kw;
      s += score[(long)b * NKV_ + idx[b * HW_ + p]];
    }
  conf[t] = s * 0.25f * INV_CNT;
}

// ---------------- SR conv as GEMM: [M_,1024] x [1024,256] per batch ----------------
__global__ void k_conv_gemm(const bf16* __restrict__ imcol, const bf16* __restrict__ wt,
                            float* __restrict__ out) {
  int id = blockIdx.x * (blockDim.x >> 5) + (threadIdx.x >> 5);  // B*64*4 waves
  int nt = id & 3, mt = (id >> 2) & 63, b = id >> 8;
  const bf16* A = imcol + ((long)b * M_ + mt * 16) * 1024;
  v8f acc[4] = {{}, {}, {}, {}};
  for (int k = 0; k < 1024; k += 32) {
    v16bf a = load_a16x32(A + k, 1024);
    v16bf bb[4];
    #pragma unroll
    for (int j = 0; j < 4; ++j)
      bb[j] = load_bT16(wt + (nt * 64 + j * 16) * 1024 + k, 1024);
    #pragma unroll
    for (int j = 0; j < 4; ++j) acc[j] = wmma_bf16(a, bb[j], acc[j]);
  }
  int l = lane_id(); int col = l & 15; int rb = (l & 16) ? 8 : 0;
  float* o = out + ((long)b * M_ + mt * 16) * 256 + nt * 64;
  #pragma unroll
  for (int j = 0; j < 4; ++j)
    #pragma unroll
    for (int r = 0; r < 8; ++r)
      o[(rb + r) * 256 + j * 16 + col] = acc[j][r];
}

// ---------------- LayerNorm(conv + sr_b) -> bf16, wave per row ----------------
__global__ void k_ln(const float* __restrict__ x, const float* __restrict__ srb,
                     const float* __restrict__ lnw, const float* __restrict__ lnb,
                     bf16* __restrict__ out) {
  int row = blockIdx.x * (blockDim.x >> 5) + (threadIdx.x >> 5);  // B*M_ rows
  int l = lane_id();
  const float* xr = x + (long)row * 256;
  float v[8]; float s = 0.f;
  #pragma unroll
  for (int i = 0; i < 8; ++i) { v[i] = xr[l * 8 + i] + srb[l * 8 + i]; s += v[i]; }
  #pragma unroll
  for (int m = 16; m; m >>= 1) s += __shfl_xor(s, m, 32);
  float mu = s * (1.f / 256.f);
  float q = 0.f;
  #pragma unroll
  for (int i = 0; i < 8; ++i) { float d = v[i] - mu; q += d * d; }
  #pragma unroll
  for (int m = 16; m; m >>= 1) q += __shfl_xor(q, m, 32);
  float rstd = rsqrtf(q * (1.f / 256.f) + LN_EPS_);
  bf16* orow = out + (long)row * 256;
  #pragma unroll
  for (int i = 0; i < 8; ++i) {
    int c = l * 8 + i;
    orow[c] = (bf16)((v[i] - mu) * rstd * lnw[c] + lnb[c]);
  }
}

// ---------------- Q projection: q = (q_x @ q_w^T) * SCALE -> [bh][Nq][HD] bf16 ----------------
__global__ void k_qproj(const bf16* __restrict__ qx, const bf16* __restrict__ qw,
                        bf16* __restrict__ qout) {
  int id = blockIdx.x * (blockDim.x >> 5) + (threadIdx.x >> 5);  // 1024*4 waves
  int nt = id & 3, mt = id >> 2;
  const bf16* A = qx + (long)mt * 16 * 256;
  v8f acc[4] = {{}, {}, {}, {}};
  for (int k = 0; k < 256; k += 32) {
    v16bf a = load_a16x32(A + k, 256);
    v16bf bb[4];
    #pragma unroll
    for (int j = 0; j < 4; ++j)
      bb[j] = load_bT16(qw + (nt * 64 + j * 16) * 256 + k, 256);
    #pragma unroll
    for (int j = 0; j < 4; ++j) acc[j] = wmma_bf16(a, bb[j], acc[j]);
  }
  int l = lane_id(); int col = l & 15; int rb = (l & 16) ? 8 : 0;
  #pragma unroll
  for (int j = 0; j < 4; ++j) {
    int c = nt * 64 + j * 16 + col;
    int h = c >> 5, dd = c & 31;
    #pragma unroll
    for (int r = 0; r < 8; ++r) {
      int m = mt * 16 + rb + r;
      int b = m >> 11, qi = m & 2047;
      qout[(((long)(b * 8 + h)) * 2048 + qi) * 32 + dd] = (bf16)(acc[j][r] * SCALE_f);
    }
  }
}

// ---------------- KV projection: kv_ln @ kv_w^T -> K [bh][M][HD], V^T [bh][HD][M] ----------------
__global__ void k_kvproj(const bf16* __restrict__ kvln, const bf16* __restrict__ kvw,
                         bf16* __restrict__ kout, bf16* __restrict__ vtout) {
  int id = blockIdx.x * (blockDim.x >> 5) + (threadIdx.x >> 5);  // 512*8 waves
  int nt = id & 7, mt = id >> 3;
  const bf16* A = kvln + (long)mt * 16 * 256;
  v8f acc[4] = {{}, {}, {}, {}};
  for (int k = 0; k < 256; k += 32) {
    v16bf a = load_a16x32(A + k, 256);
    v16bf bb[4];
    #pragma unroll
    for (int j = 0; j < 4; ++j)
      bb[j] = load_bT16(kvw + (nt * 64 + j * 16) * 256 + k, 256);
    #pragma unroll
    for (int j = 0; j < 4; ++j) acc[j] = wmma_bf16(a, bb[j], acc[j]);
  }
  int l = lane_id(); int col = l & 15; int rb = (l & 16) ? 8 : 0;
  #pragma unroll
  for (int j = 0; j < 4; ++j) {
    int c = nt * 64 + j * 16 + col;
    int kv = c >> 8, h = (c >> 5) & 7, dd = c & 31;
    #pragma unroll
    for (int r = 0; r < 8; ++r) {
      int m = mt * 16 + rb + r;
      int b = m >> 10, mm = m & 1023;
      if (kv == 0)
        kout[(((long)(b * 8 + h)) * 1024 + mm) * 32 + dd] = (bf16)acc[j][r];
      else
        vtout[(((long)(b * 8 + h)) * 32 + dd) * 1024 + mm] = (bf16)acc[j][r];
    }
  }
}

// ---------------- attention ----------------
// Block = 128 threads (4 waves) handling one (b,h) and 64 q-rows (16 per wave).
// K tiles (32 tok x 32 hd, contiguous 2KB) staged to LDS with async-to-LDS loads
// (one instruction per wave, 512B each). V tiles (32 hd-rows x 64B, stride 2KB)
// staged per-wave with a TDM tensor_load_to_lds 2-D descriptor (8 rows each).
// Double-buffered: wait asynccnt/tensorcnt -> barrier -> kick next DMA -> compute.
__global__ void __launch_bounds__(128) k_attn(const bf16* __restrict__ qb,
                                              const bf16* __restrict__ kb,
                                              const bf16* __restrict__ vtb,
                                              const float* __restrict__ conf,
                                              bf16* __restrict__ xout) {
  __shared__ bf16 kbuf[2][32 * 32];
  __shared__ bf16 vbuf[2][32 * 32];
  __shared__ bf16 pshm[4][16 * 32];
  int tid = threadIdx.x;
  int w   = tid >> 5;
  int id  = blockIdx.x;                        // B*NH*32
  int qg  = id & 31, bh = id >> 5, b = bh >> 3;
  int qt  = qg * 4 + w;
  int l = lane_id(); int col = l & 15; int rb = (l & 16) ? 8 : 0;

  const bf16* kptr = kb  + (long)bh * 1024 * 32;
  const bf16* vptr = vtb + (long)bh * 32 * 1024;
  const float* cf  = conf + b * 1024;

  v16bf aq = load_a16x32(qb + ((long)bh * 2048 + qt * 16) * 32, 32);  // K=32 == HD

  float rmax[8], rsum[8];
  v8f o0 = {}, o1 = {};
  v8f z = {};
  #pragma unroll
  for (int r = 0; r < 8; ++r) { rmax[r] = -3.0e38f; rsum[r] = 0.f; }
  bf16* sp = pshm[w];

  // ---- DMA staging of one 32-token tile (token start ts) into buffer `buf` ----
  auto stage = [&](int ts, int buf) {
    // K tile: 2KB contiguous; 128 threads x 16B via async-to-LDS (1 instr/wave).
    unsigned int klds = (unsigned int)(unsigned long)(&kbuf[buf][0]) + tid * 16;
    unsigned int kvo  = (unsigned int)tid * 16;
    unsigned long ksa = (unsigned long)(kptr + (long)ts * 32);
    asm volatile("global_load_async_to_lds_b128 %0, %1, %2"
                 :: "v"(klds), "v"(kvo), "s"(ksa) : "memory");
    // V tile: per-wave TDM 2-D descriptor, 8 hd-rows x 32 tokens, stride 1024 elems.
    unsigned long ga = (unsigned long)(vptr + (long)(w * 8) * 1024 + ts);
    su4 g0;
    g0.x = 1u;                                        // count=1, user descriptor
    g0.y = (unsigned int)(unsigned long)(&vbuf[buf][0]) + (unsigned int)(w * 8 * 64);
    g0.z = (unsigned int)ga;                          // global_addr[31:0]
    g0.w = (unsigned int)(ga >> 32) | (2u << 30);     // global_addr[56:32] | type=2
    su8 g1;
    g1[0] = (1u << 16);          // data_size = 1 (2-byte elements)
    g1[1] = (1024u << 16);       // tensor_dim0[15:0] = 1024 (bits 79:48 low half)
    g1[2] = (32u << 16);         // tensor_dim0 hi=0 | tensor_dim1[15:0] = 32
    g1[3] = (32u << 16);         // tensor_dim1 hi=0 | tile_dim0 = 32
    g1[4] = 8u;                  // tile_dim1 = 8, tile_dim2 = 0
    g1[5] = 1024u;               // tensor_dim0_stride[31:0]
    g1[6] = 0u;                  // stride hi, dim1_stride (unused for 2-D)
    g1[7] = 0u;
    asm volatile("tensor_load_to_lds %0, %1" :: "s"(g0), "s"(g1) : "memory");
  };

  stage(0, 0);
  for (int it = 0; it < 32; ++it) {
    int cur = it & 1;
    int t = it * 32;
    asm volatile("s_wait_asynccnt 0x0" ::: "memory");
    __builtin_amdgcn_s_wait_tensorcnt(0);
    __syncthreads();                       // staged tile visible to all 4 waves
    if (it + 1 < 32) stage((it + 1) * 32, cur ^ 1);

    const bf16* kt = kbuf[cur];
    const bf16* vt = vbuf[cur];
    v8f s0 = wmma_bf16(aq, load_bT16(kt, 32), z);
    v8f s1 = wmma_bf16(aq, load_bT16(kt + 16 * 32, 32), z);
    float c0 = cf[t + col], c1 = cf[t + 16 + col];
    #pragma unroll
    for (int r = 0; r < 8; ++r) { s0[r] += c0; s1[r] += c1; }
    #pragma unroll
    for (int r = 0; r < 8; ++r) {
      float mx = fmaxf(s0[r], s1[r]);
      #pragma unroll
      for (int m = 1; m < 16; m <<= 1) mx = fmaxf(mx, __shfl_xor(mx, m, 32));
      float nm = fmaxf(rmax[r], mx);
      float sc = __expf(rmax[r] - nm);
      float p0 = __expf(s0[r] - nm), p1 = __expf(s1[r] - nm);
      float ps = p0 + p1;
      #pragma unroll
      for (int m = 1; m < 16; m <<= 1) ps += __shfl_xor(ps, m, 32);
      rsum[r] = rsum[r] * sc + ps;
      rmax[r] = nm;
      o0[r] *= sc; o1[r] *= sc;
      sp[(rb + r) * 32 + col]      = (bf16)p0;   // P tile to LDS (row-major 16x32)
      sp[(rb + r) * 32 + 16 + col] = (bf16)p1;
    }
    asm volatile("s_wait_dscnt 0x0" ::: "memory");
    v16bf ap = load_a16x32(sp, 32);              // re-fragment P as WMMA A
    o0 = wmma_bf16(ap, load_bT16(vt, 32), o0);
    o1 = wmma_bf16(ap, load_bT16(vt + 16 * 32, 32), o1);
  }
  int hc = (bh & 7) * 32;
  #pragma unroll
  for (int r = 0; r < 8; ++r) {
    float inv = 1.f / rsum[r];
    long row = (long)b * 2048 + qt * 16 + rb + r;
    xout[row * 256 + hc + col]      = (bf16)(o0[r] * inv);
    xout[row * 256 + hc + 16 + col] = (bf16)(o1[r] * inv);
  }
}

// ---------------- output projection + bias -> f32 ----------------
__global__ void k_oproj(const bf16* __restrict__ xb, const bf16* __restrict__ pw,
                        const float* __restrict__ pb, float* __restrict__ out) {
  int id = blockIdx.x * (blockDim.x >> 5) + (threadIdx.x >> 5);  // 1024*4 waves
  int nt = id & 3, mt = id >> 2;
  const bf16* A = xb + (long)mt * 16 * 256;
  v8f acc[4] = {{}, {}, {}, {}};
  for (int k = 0; k < 256; k += 32) {
    v16bf a = load_a16x32(A + k, 256);
    v16bf bb[4];
    #pragma unroll
    for (int j = 0; j < 4; ++j)
      bb[j] = load_bT16(pw + (nt * 64 + j * 16) * 256 + k, 256);
    #pragma unroll
    for (int j = 0; j < 4; ++j) acc[j] = wmma_bf16(a, bb[j], acc[j]);
  }
  int l = lane_id(); int col = l & 15; int rb = (l & 16) ? 8 : 0;
  float* o = out + (long)mt * 16 * 256 + nt * 64;
  #pragma unroll
  for (int j = 0; j < 4; ++j) {
    float bias = pb[nt * 64 + j * 16 + col];
    #pragma unroll
    for (int r = 0; r < 8; ++r)
      o[(rb + r) * 256 + j * 16 + col] = acc[j][r] + bias;
  }
}

extern "C" void kernel_launch(void* const* d_in, const int* in_sizes, int n_in,
                              void* d_out, int out_size, void* d_ws, size_t ws_size,
                              hipStream_t stream) {
  (void)in_sizes; (void)n_in; (void)out_size; (void)ws_size;
  const float* q_x         = (const float*)d_in[0];
  const float* kv_x        = (const float*)d_in[1];
  const float* token_score = (const float*)d_in[2];
  const int*   idx_token   = (const int*)d_in[3];
  const float* q_w         = (const float*)d_in[4];
  const float* kv_w        = (const float*)d_in[5];
  const float* proj_w      = (const float*)d_in[6];
  const float* proj_b      = (const float*)d_in[7];
  const float* sr_w        = (const float*)d_in[8];
  const float* sr_b        = (const float*)d_in[9];
  const float* ln_w        = (const float*)d_in[10];
  const float* ln_b        = (const float*)d_in[11];

  char* ws = (char*)d_ws;
  size_t off = 0;
  bf16* qx_bf   = (bf16*)(ws + off); off += (size_t)B_ * NQ_ * C_ * 2;      //  8 MB
  bf16* qw_bf   = (bf16*)(ws + off); off += (size_t)C_ * C_ * 2;            // 128 KB
  bf16* kvw_bf  = (bf16*)(ws + off); off += (size_t)2 * C_ * C_ * 2;        // 256 KB
  bf16* pw_bf   = (bf16*)(ws + off); off += (size_t)C_ * C_ * 2;            // 128 KB
  bf16* wconv   = (bf16*)(ws + off); off += (size_t)256 * 1024 * 2;         // 512 KB
  bf16* imcol   = (bf16*)(ws + off); off += (size_t)B_ * M_ * 1024 * 2;     // 16 MB
  float* conf   = (float*)(ws + off); off += (size_t)B_ * M_ * 4;           //  32 KB
  float* convo  = (float*)(ws + off); off += (size_t)B_ * M_ * C_ * 4;      //  8 MB
  bf16* kvln    = (bf16*)(ws + off); off += (size_t)B_ * M_ * C_ * 2;       //  4 MB
  bf16* k_bf    = (bf16*)(ws + off); off += (size_t)B_ * NH_ * M_ * HD_ * 2;//  4 MB
  bf16* vt_bf   = (bf16*)(ws + off); off += (size_t)B_ * NH_ * HD_ * M_ * 2;//  4 MB
  bf16* q_bf    = (bf16*)(ws + off); off += (size_t)B_ * NH_ * NQ_ * HD_ * 2;// 8 MB
  bf16* x_bf    = (bf16*)(ws + off); off += (size_t)B_ * NQ_ * C_ * 2;      //  8 MB

  float* out = (float*)d_out;

  // 1) precision conversion / packing
  int nqx = B_ * NQ_ * C_;
  k_cvt_bf16<<<(nqx + 255) / 256, 256, 0, stream>>>(q_x, qx_bf, nqx);
  k_cvt_bf16<<<(C_ * C_ + 255) / 256, 256, 0, stream>>>(q_w, qw_bf, C_ * C_);
  k_cvt_bf16<<<(2 * C_ * C_ + 255) / 256, 256, 0, stream>>>(kv_w, kvw_bf, 2 * C_ * C_);
  k_cvt_bf16<<<(C_ * C_ + 255) / 256, 256, 0, stream>>>(proj_w, pw_bf, C_ * C_);
  k_pack_convw<<<(256 * 1024) / 256, 256, 0, stream>>>(sr_w, wconv);

  // 2) token2map gather fused with im2col; conf pooling
  k_im2col<<<(B_ * HW_ * 64) / 256, 256, 0, stream>>>(kv_x, idx_token, imcol);
  k_conf<<<(B_ * M_) / 256, 256, 0, stream>>>(token_score, idx_token, conf);

  // 3) SR conv as WMMA GEMM, then LayerNorm
  k_conv_gemm<<<(B_ * 64 * 4) / 8, 256, 0, stream>>>(imcol, wconv, convo);
  k_ln<<<(B_ * M_) / 8, 256, 0, stream>>>(convo, sr_b, ln_w, ln_b, kvln);

  // 4) projections
  k_qproj<<<(1024 * 4) / 8, 256, 0, stream>>>(qx_bf, qw_bf, q_bf);
  k_kvproj<<<(512 * 8) / 8, 256, 0, stream>>>(kvln, kvw_bf, k_bf, vt_bf);

  // 5) attention (async/TDM-staged K/V in LDS)
  k_attn<<<B_ * NH_ * 32, 128, 0, stream>>>(q_bf, k_bf, vt_bf, conf, x_bf);

  // 6) output projection
  k_oproj<<<(1024 * 4) / 8, 256, 0, stream>>>(x_bf, pw_bf, proj_b, out);
}
